// Generator_20555713478694
// MI455X (gfx1250) — compile-verified, matrix operand order checked by reading, not run on hardware
//
#include <hip/hip_runtime.h>
#include <hip/hip_bf16.h>
#include <math.h>

typedef __attribute__((ext_vector_type(16))) _Float16 v16h;
typedef __attribute__((ext_vector_type(8)))  float    v8f;

#define EPSV 1e-5f

// ---------------- noise linear: out[768] = z[512] @ W^T + b ----------------
__global__ void k_linear(const float* __restrict__ z, const float* __restrict__ W,
                         const float* __restrict__ b, float* __restrict__ out) {
    int o = blockIdx.x * blockDim.x + threadIdx.x;
    if (o >= 768) return;
    const float* w = W + (long)o * 512;
    float acc = b[o];
    #pragma unroll 4
    for (int i = 0; i < 512; ++i) acc += z[i] * w[i];
    out[o] = acc;
}

// ---- fused: both 5x5 valid convs + nearest resize to R, NCHW, C=3 ----
__global__ void k_conv2_up(const float* __restrict__ real,
                           const float* __restrict__ w1, const float* __restrict__ b1,
                           const float* __restrict__ w2, const float* __restrict__ b2,
                           int N, int Hin, int Win, int R,
                           float* __restrict__ y, float* __restrict__ y2) {
    __shared__ float sw1[225], sw2[225], sb1[3], sb2[3];
    int t = threadIdx.x;
    if (t < 225) { sw1[t] = w1[t]; sw2[t] = w2[t]; }
    if (t < 3)   { sb1[t] = b1[t]; sb2[t] = b2[t]; }
    __syncthreads();

    int Hc = Hin - 4, Wc = Win - 4;
    long total = (long)N * R * R;
    long idx = (long)blockIdx.x * blockDim.x + t;
    if (idx >= total) return;

    int ox = (int)(idx % R);
    long tmp = idx / R;
    int oy = (int)(tmp % R);
    int n  = (int)(tmp / R);
    int sy = (oy * Hc) / R, sx = (ox * Wc) / R;   // nearest: floor(dst*in/out)

    const float* base = real + (long)n * 3 * Hin * Win;
    float a0 = sb1[0], a1 = sb1[1], a2 = sb1[2];
    float c0 = sb2[0], c1 = sb2[1], c2 = sb2[2];
    #pragma unroll
    for (int ci = 0; ci < 3; ++ci) {
        const float* ip = base + (long)ci * Hin * Win + (long)sy * Win + sx;
        #pragma unroll
        for (int ky = 0; ky < 5; ++ky) {
            const float* row = ip + (long)ky * Win;
            #pragma unroll
            for (int kx = 0; kx < 5; ++kx) {
                float v = row[kx];
                int wi = ci * 25 + ky * 5 + kx;
                a0 += v * sw1[wi];       a1 += v * sw1[75 + wi];  a2 += v * sw1[150 + wi];
                c0 += v * sw2[wi];       c1 += v * sw2[75 + wi];  c2 += v * sw2[150 + wi];
            }
        }
    }
    long HWr = (long)R * R;
    long o = (long)n * 3 * HWr + (long)oy * R + ox;
    y[o]  = a0; y[o + HWr]  = a1; y[o + 2 * HWr]  = a2;
    y2[o] = c0; y2[o + HWr] = c1; y2[o + 2 * HWr] = c2;
}

// ---------------- per-(n,c) sum / sumsq over HW ----------------
__global__ void k_stats(const float* __restrict__ y, int HW,
                        float* __restrict__ s_out, float* __restrict__ q_out) {
    int g = blockIdx.x;                  // n*3 + c
    const float* p = y + (long)g * HW;
    float s = 0.f, q = 0.f;
    for (int i = threadIdx.x; i < HW; i += blockDim.x) { float v = p[i]; s += v; q += v * v; }
    __shared__ float ss[512], sq[512];
    ss[threadIdx.x] = s; sq[threadIdx.x] = q; __syncthreads();
    for (int off = blockDim.x >> 1; off > 0; off >>= 1) {
        if (threadIdx.x < (unsigned)off) { ss[threadIdx.x] += ss[threadIdx.x + off]; sq[threadIdx.x] += sq[threadIdx.x + off]; }
        __syncthreads();
    }
    if (threadIdx.x == 0) { s_out[g] = ss[0]; q_out[g] = sq[0]; }
}

// -------- fold instance+batch norms + affine + p1 into (c0, c1) per (n,c) --------
__global__ void k_finalize(const float* __restrict__ s_in, const float* __restrict__ q_in,
                           int N, int HW,
                           const float* __restrict__ in_p, const float* __restrict__ ln_p,
                           const float* __restrict__ bn_p, const float* __restrict__ bn_w,
                           const float* __restrict__ bn_b, const float* __restrict__ p1p,
                           float* __restrict__ c0_out, float* __restrict__ c1_out) {
    __shared__ float bm[3], bv[3];
    int t = threadIdx.x;
    if (t < 3) {
        float s = 0.f, q = 0.f;
        for (int n = 0; n < N; ++n) { s += s_in[n * 3 + t]; q += q_in[n * 3 + t]; }
        float cnt = (float)N * (float)HW;
        float m = s / cnt;
        bm[t] = m; bv[t] = q / cnt - m * m;
    }
    __syncthreads();
    if (t < N * 3) {
        int c = t % 3;
        float mi = s_in[t] / (float)HW;
        float vi = q_in[t] / (float)HW - mi * mi;
        float A = in_p[0] + in_p[1] + in_p[2] + ln_p[0] + ln_p[1] + ln_p[2];
        float B = bn_p[0] + bn_p[1] + bn_p[2];
        float alpha = A * rsqrtf(vi + EPSV);
        float beta  = B * bn_w[c] * rsqrtf(bv[c] + EPSV);
        float p1 = p1p[0];
        c0_out[t] = (1.f + alpha + beta) * p1;
        c1_out[t] = (-alpha * mi - beta * bm[c] + B * bn_b[c]) * p1;
    }
}

// -------- elementwise: dst = c0*y + c1 + p2*y2 (+noise) (tanh) --------
__global__ void k_apply(const float* __restrict__ y, const float* __restrict__ y2,
                        const float* __restrict__ c0, const float* __restrict__ c1,
                        const float* __restrict__ p2p, const float* __restrict__ noise,
                        int HW, long total, int do_tanh, float* __restrict__ dst) {
    long idx = (long)blockIdx.x * blockDim.x + threadIdx.x;
    if (idx >= total) return;
    long nc = idx / HW;
    float v = c0[nc] * y[idx] + c1[nc] + p2p[0] * y2[idx];
    if (noise) v += noise[idx % (3L * HW)];
    if (do_tanh) v = tanhf(v);
    dst[idx] = v;
}

// ---------------- attention: q/k/v = 3x3 channel mix per pixel ----------------
__global__ void k_qkv(const float* __restrict__ h,
                      const float* __restrict__ qw, const float* __restrict__ qb,
                      const float* __restrict__ kw, const float* __restrict__ kb,
                      const float* __restrict__ vw, const float* __restrict__ vb,
                      int HW, int total,
                      float* __restrict__ q, float* __restrict__ k, float* __restrict__ v) {
    int idx = blockIdx.x * blockDim.x + threadIdx.x;
    if (idx >= total) return;
    int b = idx / HW, p = idx % HW;
    const float* hp = h + (long)b * 3 * HW + p;
    float x0 = hp[0], x1 = hp[HW], x2 = hp[2 * HW];
    long o = (long)b * 3 * HW + p;
    q[o]          = qw[0] * x0 + qw[1] * x1 + qw[2] * x2 + qb[0];
    q[o + HW]     = qw[3] * x0 + qw[4] * x1 + qw[5] * x2 + qb[1];
    q[o + 2 * HW] = qw[6] * x0 + qw[7] * x1 + qw[8] * x2 + qb[2];
    k[o]          = kw[0] * x0 + kw[1] * x1 + kw[2] * x2 + kb[0];
    k[o + HW]     = kw[3] * x0 + kw[4] * x1 + kw[5] * x2 + kb[1];
    k[o + 2 * HW] = kw[6] * x0 + kw[7] * x1 + kw[8] * x2 + kb[2];
    v[o]          = vw[0] * x0 + vw[1] * x1 + vw[2] * x2 + vb[0];
    v[o + HW]     = vw[3] * x0 + vw[4] * x1 + vw[5] * x2 + vb[1];
    v[o + 2 * HW] = vw[6] * x0 + vw[7] * x1 + vw[8] * x2 + vb[2];
}

// -------- per-row softmax stats: mx[m], and vtilde = v/Z packed f16 --------
__global__ void k_rowstats(const float* __restrict__ q, const float* __restrict__ k,
                           const float* __restrict__ v, int HW, int total,
                           float* __restrict__ mx, _Float16* __restrict__ vt) {
    int idx = blockIdx.x * blockDim.x + threadIdx.x;
    if (idx >= total) return;
    int b = idx / HW, m = idx % HW;
    const float* qb_ = q + (long)b * 3 * HW;
    const float* kp  = k + (long)b * 3 * HW + m;
    float k0 = kp[0], k1 = kp[HW], k2 = kp[2 * HW];
    float mmax = -1e30f;
    for (int n = 0; n < HW; ++n) {
        float s = k0 * qb_[n] + k1 * qb_[n + HW] + k2 * qb_[n + 2 * HW];
        mmax = fmaxf(mmax, s);
    }
    float Z = 0.f;
    for (int n = 0; n < HW; ++n) {
        float s = k0 * qb_[n] + k1 * qb_[n + HW] + k2 * qb_[n + 2 * HW];
        Z += __expf(s - mmax);
    }
    mx[idx] = mmax;
    float iz = 1.f / Z;
    const float* vp = v + (long)b * 3 * HW + m;
    _Float16* vtp = vt + (long)b * 3 * HW + m;
    vtp[0]      = (_Float16)(vp[0] * iz);
    vtp[HW]     = (_Float16)(vp[HW] * iz);
    vtp[2 * HW] = (_Float16)(vp[2 * HW] * iz);
}

// -------- attention output GEMM via V_WMMA_F32_16X16X32_F16, residual in place --------
// One wave per (batch, 16-column n-tile). K dim = m (HW), stepped 32 at a time.
// A-fragment: lane l holds row M=l%16, K range (l&16)?16..31:0..15 -> vtilde (rows>=3 zero).
// B-fragment: lane l holds column N=l%16, same K half -> each lane computes its own
// 16 exp-weights locally (no cross-lane traffic needed).
__global__ void __launch_bounds__(32)
k_attn_out(const float* __restrict__ q, const float* __restrict__ k,
           const float* __restrict__ mx, const _Float16* __restrict__ vt,
           int HW, float* __restrict__ h) {
    int ntiles = HW >> 4;
    int b  = blockIdx.x / ntiles;
    int nt = blockIdx.x % ntiles;
    int l = threadIdx.x;
    int lane16 = l & 15;
    int khalf  = (l & 16);                 // 0 or 16
    int n = (nt << 4) + lane16;

    const float*    qb_ = q  + (long)b * 3 * HW;
    const float*    kb_ = k  + (long)b * 3 * HW;
    const float*    mxb = mx + (long)b * HW;
    const _Float16* vtb = vt + (long)b * 3 * HW;

    float q0 = qb_[n], q1 = qb_[n + HW], q2 = qb_[n + 2 * HW];
    v8f acc = {};
    for (int mc = 0; mc < HW; mc += 32) {
        int mb = mc + khalf;
        v16h bfrag;
        #pragma unroll
        for (int j = 0; j < 16; ++j) {
            int m = mb + j;
            float s = kb_[m] * q0 + kb_[m + HW] * q1 + kb_[m + 2 * HW] * q2;
            bfrag[j] = (_Float16)__expf(s - mxb[m]);
        }
        v16h afrag = {};
        if (lane16 < 3) {
            const _Float16* ap = vtb + (long)lane16 * HW + mb;
            #pragma unroll
            for (int j = 0; j < 16; ++j) afrag[j] = ap[j];
        }
        acc = __builtin_amdgcn_wmma_f32_16x16x32_f16(false, afrag, false, bfrag,
                                                     (short)0, acc, false, false);
    }
    if (l < 16) {   // D layout: lanes 0..15, VGPR v -> M=v, N=lane
        long o = (long)b * 3 * HW + n;
        h[o]          += acc[0];
        h[o + HW]     += acc[1];
        h[o + 2 * HW] += acc[2];
    }
}

extern "C" void kernel_launch(void* const* d_in, const int* in_sizes, int n_in,
                              void* d_out, int out_size, void* d_ws, size_t ws_size,
                              hipStream_t stream) {
    const float* x      = (const float*)d_in[0];
    const float* zin    = (const float*)d_in[1];
    const float* lin_w  = (const float*)d_in[2];
    const float* lin_b  = (const float*)d_in[3];
    const float* conv_w = (const float*)d_in[4];
    const float* conv_b = (const float*)d_in[5];
    const float* conv2w = (const float*)d_in[6];
    const float* conv2b = (const float*)d_in[7];
    const float* q_w = (const float*)d_in[8];
    const float* q_b = (const float*)d_in[9];
    const float* k_w = (const float*)d_in[10];
    const float* k_b = (const float*)d_in[11];
    const float* v_w = (const float*)d_in[12];
    const float* v_b = (const float*)d_in[13];
    const float* in_p = (const float*)d_in[14];
    const float* ln_p = (const float*)d_in[15];
    const float* bn_p = (const float*)d_in[16];
    const float* bn_w = (const float*)d_in[17];
    const float* bn_b = (const float*)d_in[18];
    const float* p1   = (const float*)d_in[19];
    const float* p2   = (const float*)d_in[20];
    float* out = (float*)d_out;

    // workspace carve-up (needs ~53 MB; everything L2-resident on MI455X)
    char* ws = (char*)d_ws;
    size_t off = 0;
    auto alloc = [&](size_t bytes) -> void* {
        void* p = ws + off;
        off = (off + bytes + 255) & ~(size_t)255;
        return p;
    };
    const long NMAX = 32L * 3 * 256 * 256;
    float* big0 = (float*)alloc(NMAX * 4);
    float* big1 = (float*)alloc(NMAX * 4);
    float* qb   = (float*)alloc(32L * 3 * 1024 * 4);
    float* kb   = (float*)alloc(32L * 3 * 1024 * 4);
    float* vb   = (float*)alloc(32L * 3 * 1024 * 4);
    float* mxb  = (float*)alloc(32L * 1024 * 4);
    _Float16* vtb = (_Float16*)alloc(32L * 3 * 1024 * 2);
    float* inst_s = (float*)alloc(96 * 4);
    float* inst_q = (float*)alloc(96 * 4);
    float* c0b    = (float*)alloc(96 * 4);
    float* c1b    = (float*)alloc(96 * 4);
    float* nb0 = (float*)alloc(3L * 64 * 64 * 4);
    float* nb1 = (float*)alloc(3L * 64 * 64 * 4);
    float* nb2 = (float*)alloc(3L * 64 * 64 * 4);

    auto run_block = [&](int i, const float* cur, int N, int Hin, int R,
                         float* ybuf, float* y2buf, float* dst,
                         const float* noise, int do_tanh, int do_attn) {
        int HW = R * R;
        long ctotal = (long)N * HW;
        k_conv2_up<<<(unsigned)((ctotal + 255) / 256), 256, 0, stream>>>(
            cur, conv_w + (long)i * 225, conv_b + (long)i * 3,
            conv2w + (long)i * 225, conv2b + (long)i * 3,
            N, Hin, Hin, R, ybuf, y2buf);
        k_stats<<<N * 3, 512, 0, stream>>>(ybuf, HW, inst_s, inst_q);
        k_finalize<<<1, 128, 0, stream>>>(inst_s, inst_q, N, HW,
            in_p + (long)i * 3, ln_p + (long)i * 3, bn_p + (long)i * 3,
            bn_w + (long)i * 3, bn_b + (long)i * 3, p1 + i, c0b, c1b);
        long etotal = (long)N * 3 * HW;
        k_apply<<<(unsigned)((etotal + 255) / 256), 256, 0, stream>>>(
            ybuf, y2buf, c0b, c1b, p2 + i, noise, HW, etotal, do_tanh, dst);
        if (do_attn) {
            int atotal = N * HW;
            k_qkv<<<(atotal + 255) / 256, 256, 0, stream>>>(
                dst, q_w + (long)i * 9, q_b + (long)i * 3,
                k_w + (long)i * 9, k_b + (long)i * 3,
                v_w + (long)i * 9, v_b + (long)i * 3,
                HW, atotal, qb, kb, vb);
            k_rowstats<<<(atotal + 255) / 256, 256, 0, stream>>>(
                qb, kb, vb, HW, atotal, mxb, vtb);
            k_attn_out<<<N * (HW / 16), 32, 0, stream>>>(qb, kb, mxb, vtb, HW, dst);
        }
    };

    // ---- noise path: Linear(512->768) -> (1,3,16,16) -> block0(r=32) -> block1(r=64) ----
    k_linear<<<3, 256, 0, stream>>>(zin, lin_w, lin_b, nb0);
    run_block(0, nb0, 1, 16, 32, nb1, nb2, nb2, nullptr, 0, 0);   // -> nb2 (1,3,32,32)
    run_block(1, nb2, 1, 32, 64, nb1, nb0, nb0, nullptr, 0, 0);   // -> nb0 (1,3,64,64)

    // ---- main path; buffers rotate through {big0, big1, out}; attn is in-place ----
    run_block(2,  x,    32, 256, 256, big0, big1, big1, nullptr, 0, 0);
    run_block(3,  big1, 32, 256, 128, big0, out,  out,  nullptr, 0, 0);
    run_block(4,  out,  32, 128, 64,  big0, big1, big1, nb0,     0, 0);  // + noise
    run_block(5,  big1, 32, 64,  32,  big0, out,  out,  nullptr, 0, 1);
    run_block(6,  out,  32, 32,  16,  big0, big1, big1, nullptr, 0, 1);
    run_block(7,  big1, 32, 16,  16,  big0, out,  out,  nullptr, 0, 1);
    run_block(8,  out,  32, 16,  32,  big0, big1, big1, nullptr, 0, 1);
    run_block(9,  big1, 32, 32,  64,  big0, out,  out,  nullptr, 0, 0);
    run_block(10, out,  32, 64,  128, big0, big1, big1, nullptr, 0, 0);
    run_block(11, big1, 32, 128, 256, big0, out,  out,  nullptr, 1, 0);  // tanh -> d_out
}
